// LinearQuant_48687749267956
// MI455X (gfx1250) — matrix operand order repaired and checked by reading, not compiled, not run on hardware
//
#include <hip/hip_runtime.h>

// ---------------------------------------------------------------------------
// out[32,8192] = x[32,8192] @ (int8(W)[8192,8192] * scaler)
//
// HBM-bound: 256 MB of int32-stored weights => ~11 us floor @ 23.3 TB/s.
// Strategy: bf16 WMMA (weights exact in bf16), x split into hi/lo bf16 for
// ~f32 accuracy. A-fragments are pre-swizzled into d_ws by a cheap pre-pass
// so the hot loop is loads + int->bf16 convert + 4x v_wmma per K-step.
// ---------------------------------------------------------------------------

typedef __attribute__((ext_vector_type(16))) __bf16          v16bf;
typedef __attribute__((ext_vector_type(16))) unsigned short  v16us;
typedef __attribute__((ext_vector_type(8)))  unsigned        v8u;
typedef __attribute__((ext_vector_type(8)))  float           v8f;

#define IN_F   8192
#define OUT_F  8192
#define BATCH  32
#define NWAVES 4                     // K-split factor (waves per block)
#define KCHUNK (IN_F / NWAVES)       // 2048
#define KSTEPS (IN_F / 32)           // 256 K-steps of 32

static __device__ __forceinline__ unsigned f2u(float f) {
    union { float f; unsigned u; } c; c.f = f; return c.u;
}
static __device__ __forceinline__ float u2f(unsigned u) {
    union { float f; unsigned u; } c; c.u = u; return c.f;
}
// round-to-nearest-even f32 -> bf16 bits
static __device__ __forceinline__ unsigned short bf16_rne(float f) {
    unsigned u = f2u(f);
    u += 0x7FFFu + ((u >> 16) & 1u);
    return (unsigned short)(u >> 16);
}

// ---------------------------------------------------------------------------
// Pre-pass: build pre-swizzled bf16 A-fragments (hi/lo split of x) in d_ws.
// Fragment layout (contiguous K order used by this code, 16-bit A 16x32): lane group g = lane>>4,
// row M = lane&15, half h: K = 16*g + h.
// ws layout: frag f = (kstep*2 + mtile)*2 + split ;  32B per lane, contiguous:
//            halfword index = (f*32 + lane)*16 + h        (total 1 MB)
// ---------------------------------------------------------------------------
__global__ __launch_bounds__(128)
void build_a_frags(const float* __restrict__ x, unsigned short* __restrict__ afr) {
    const int lane = threadIdx.x & 31;
    const int wv   = blockIdx.x * (blockDim.x >> 5) + (threadIdx.x >> 5);
    const int s    = wv >> 1;        // K-step
    const int mt   = wv & 1;         // M-tile
    const int g    = lane >> 4;
    const int n    = lane & 15;

    const float* xr = x + (size_t)(mt * 16 + n) * IN_F + s * 32;
    v16us hs, ls;
#pragma unroll
    for (int h = 0; h < 16; ++h) {
        const int k = 16 * g + h;
        const float f = xr[k];
        const unsigned ub = f2u(f) & 0xFFFF0000u;    // truncate: exact hi part
        hs[h] = (unsigned short)(ub >> 16);
        ls[h] = bf16_rne(f - u2f(ub));               // residual, next ~8 bits
    }
    unsigned short* ph = afr + ((size_t)(((s * 2 + mt) * 2) + 0) * 32 + lane) * 16;
    unsigned short* pl = afr + ((size_t)(((s * 2 + mt) * 2) + 1) * 32 + lane) * 16;
    *(v16us*)ph = hs;
    *(v16us*)pl = ls;
}

// ---------------------------------------------------------------------------
// Main GEMM: one 16-col N-tile per block; 4 waves split K; LDS reduction.
// ---------------------------------------------------------------------------
__global__ __launch_bounds__(32 * NWAVES)
void linq_i8_bf16wmma(const unsigned short* __restrict__ afr,
                      const int*   __restrict__ w,
                      const float* __restrict__ scaler,
                      float*       __restrict__ out) {
    const int lane = threadIdx.x & 31;   // wave32
    const int wave = threadIdx.x >> 5;
    const int g    = lane >> 4;          // lane group
    const int n    = lane & 15;          // column within N-tile
    const int ncol = blockIdx.x * 16 + n;

    v8f acc0 = {};   // rows 0..15
    v8f acc1 = {};   // rows 16..31

    const int*   wb = w + ncol;
    const v16us* af = (const v16us*)afr;     // 32B fragments, index f*32+lane

    const int s0 = wave * (KCHUNK / 32);
    for (int s = s0; s < s0 + (KCHUNK / 32); ++s) {
        // ---- B fragment: 32x16 bf16.  lane (g,n), half h -> K = 16*g + h
        const int* wp = wb + (size_t)(s * 32 + 16 * g) * OUT_F;
        __builtin_prefetch(wp + (size_t)32 * OUT_F, 0, 0);   // next K-block
        v8u bw;
#pragma unroll
        for (int h = 0; h < 16; h += 2) {
            // non-temporal: weight stream (256 MB) must not evict L2 hot data
            const float w0 = (float)__builtin_nontemporal_load(wp + (size_t)h       * OUT_F);
            const float w1 = (float)__builtin_nontemporal_load(wp + (size_t)(h + 1) * OUT_F);
            // pack the two bf16 (= f32 high halves; exact for |w|<=127)
            bw[h >> 1] = __builtin_amdgcn_perm(f2u(w1), f2u(w0), 0x07060302u);
        }
        const v16bf bfrag = __builtin_bit_cast(v16bf, bw);

        // ---- A fragments: pre-swizzled, 32B contiguous per lane (L2-hot)
        const v16bf a0h = __builtin_bit_cast(v16bf, af[(size_t)(((s * 2 + 0) * 2) + 0) * 32 + lane]);
        const v16bf a0l = __builtin_bit_cast(v16bf, af[(size_t)(((s * 2 + 0) * 2) + 1) * 32 + lane]);
        const v16bf a1h = __builtin_bit_cast(v16bf, af[(size_t)(((s * 2 + 1) * 2) + 0) * 32 + lane]);
        const v16bf a1l = __builtin_bit_cast(v16bf, af[(size_t)(((s * 2 + 1) * 2) + 1) * 32 + lane]);

        // ---- 4x v_wmma_f32_16x16x32_bf16, B reused; alternate accs to
        //      maximize distance in each accumulate chain
        acc0 = __builtin_amdgcn_wmma_f32_16x16x32_bf16(
                   false, a0l, false, bfrag, (short)0, acc0, false, false);
        acc1 = __builtin_amdgcn_wmma_f32_16x16x32_bf16(
                   false, a1l, false, bfrag, (short)0, acc1, false, false);
        acc0 = __builtin_amdgcn_wmma_f32_16x16x32_bf16(
                   false, a0h, false, bfrag, (short)0, acc0, false, false);
        acc1 = __builtin_amdgcn_wmma_f32_16x16x32_bf16(
                   false, a1h, false, bfrag, (short)0, acc1, false, false);
    }

    // ---- cross-wave K reduction through LDS (pad to 17 to dodge bank conflicts)
    __shared__ float red[NWAVES][32][17];
#pragma unroll
    for (int r = 0; r < 8; ++r) {
        red[wave][lane][r]     = acc0[r];
        red[wave][lane][8 + r] = acc1[r];
    }
    __syncthreads();

    if (wave == 0) {
        const float s = scaler[0];
#pragma unroll
        for (int i = 0; i < 16; ++i) {
            float v = red[0][lane][i] + red[1][lane][i]
                    + red[2][lane][i] + red[3][lane][i];
            const int mt  = i >> 3;
            const int r   = i & 7;
            const int row = mt * 16 + r + 8 * g;   // C/D layout: M = r + 8*g
            out[(size_t)row * OUT_F + ncol] = v * s;
        }
    }
}

extern "C" void kernel_launch(void* const* d_in, const int* in_sizes, int n_in,
                              void* d_out, int out_size, void* d_ws, size_t ws_size,
                              hipStream_t stream) {
    (void)in_sizes; (void)n_in; (void)out_size; (void)ws_size;
    const float* x      = (const float*)d_in[0];
    const int*   wq     = (const int*)d_in[1];
    const float* scaler = (const float*)d_in[2];
    float*       out    = (float*)d_out;
    unsigned short* afr = (unsigned short*)d_ws;   // needs 1 MB scratch

    // Pre-pass: 512 fragment-pairs (256 K-steps x 2 M-tiles), one wave each.
    build_a_frags<<<dim3(128), dim3(128), 0, stream>>>(x, afr);

    // Main GEMM: one N-tile of 16 columns per block, 4 waves split K.
    linq_i8_bf16wmma<<<dim3(OUT_F / 16), dim3(32 * NWAVES), 0, stream>>>(
        afr, wq, scaler, out);
}
